// BasicAttentionModel_7121055776964
// MI455X (gfx1250) — compile-verified, matrix operand order checked by reading, not run on hardware
//
#include <hip/hip_runtime.h>
#include <math.h>

// ---------------------------------------------------------------------------
// BasicAttentionModel on MI455X (gfx1250).
// f16 activations/weights in HBM (halves 23.3 TB/s traffic), fp32 accum via
// v_wmma_f32_16x16x32_f16. Block-tiled GEMM: 128x128 C tile per 256-thread
// block, double-buffered f16 LDS tiles (one barrier / K-step), fragments as
// 2x ds_read_b128, compile-time N/K so stores use immediate offsets.
// ---------------------------------------------------------------------------

typedef __attribute__((ext_vector_type(16))) _Float16 v16h;
typedef __attribute__((ext_vector_type(8)))  _Float16 v8h;
typedef __attribute__((ext_vector_type(8)))  float    v8f;

#define D_MODEL 512
#define SEQ     1024
#define NBATCH  32
#define MROWS   (NBATCH * SEQ)   // 32768 flattened (b, s) rows

#define BM 128
#define BN 128
#define BK 32
#define LDST 40                  // LDS tile row stride in f16 (32 data + 8 pad)

// ---------------------------------------------------------------------------
// Block-tiled WMMA GEMM.  C = scale * A[M,K] x B (+bias)(+relu)
//   A: f16 row-major (ld = K).  B: f16, K x N row-major, or N x K if TRANSB.
//   Output f16 (Ch) or f32 (Cf) via OUT32. Batched via blockIdx.z.
// Wave grid 4x2 over the 128x128 tile; each wave owns 2x4 WMMA tiles.
// ---------------------------------------------------------------------------
template <int N, int K, bool TRANSB, bool BIAS, bool RELU, bool OUT32>
__global__ __launch_bounds__(256) void gemm_wmma(
    const _Float16* __restrict__ A, const _Float16* __restrict__ Bm,
    const float* __restrict__ bias, _Float16* __restrict__ Ch,
    float* __restrict__ Cf, float scale,
    long long sA, long long sB, long long sC) {
  __shared__ alignas(16) _Float16 lA[2][BM * LDST];
  __shared__ alignas(16) _Float16 lB[2][BN * LDST];   // stored [n][k]

  const int tid  = threadIdx.x;
  const int lane = tid & 31;
  const int wave = tid >> 5;
  const int half = lane >> 4;
  const int l16  = lane & 15;
  const int wm   = wave & 3;        // 4 waves along M
  const int wn   = wave >> 2;       // 2 waves along N
  const int bm0  = blockIdx.x * BM;
  const int bn0  = blockIdx.y * BN;

  const _Float16* Ab = A  + (long long)blockIdx.z * sA;
  const _Float16* Bb = Bm + (long long)blockIdx.z * sB;

  v8f acc[2][4] = {};

  // staging coordinates
  const int sr = tid >> 1;          // 0..127 : tile row (A / B^T)
  const int sh = tid & 1;           // which 16-f16 half of the 32-wide K chunk
  const int bk = tid >> 3;          // 0..31  : k row (non-transposed B)
  const int bn = (tid & 7) * 16;    // 0..112 : n chunk (non-transposed B)

  // ---- prologue: stage tile k0 = 0 into buffer 0 ------------------------
  {
    const uint4* ga = (const uint4*)(Ab + (long long)(bm0 + sr) * K + sh * 16);
    uint4 a0 = ga[0], a1 = ga[1];
    uint4* da = (uint4*)&lA[0][sr * LDST + sh * 16];
    da[0] = a0; da[1] = a1;
    if (TRANSB) {
      const uint4* gb = (const uint4*)(Bb + (long long)(bn0 + sr) * K + sh * 16);
      uint4 b0 = gb[0], b1 = gb[1];
      uint4* db = (uint4*)&lB[0][sr * LDST + sh * 16];
      db[0] = b0; db[1] = b1;
    } else {
      const uint4* gb = (const uint4*)(Bb + (long long)bk * N + bn0 + bn);
      _Float16 t[16];
      *(uint4*)&t[0] = gb[0];
      *(uint4*)&t[8] = gb[1];
#pragma unroll
      for (int j = 0; j < 16; ++j) lB[0][(bn + j) * LDST + bk] = t[j];
    }
  }
  __syncthreads();

  // ---- main pipeline: prefetch k+1 -> regs, compute buf[cur], store nxt --
  for (int k0 = 0, cur = 0; k0 < K; k0 += BK, cur ^= 1) {
    const bool has = (k0 + BK) < K;      // block-uniform
    uint4 ra0, ra1, rb0, rb1;
    if (has) {
      const int kn = k0 + BK;
      const uint4* ga = (const uint4*)(Ab + (long long)(bm0 + sr) * K + kn + sh * 16);
      ra0 = ga[0]; ra1 = ga[1];
      if (TRANSB) {
        const uint4* gb = (const uint4*)(Bb + (long long)(bn0 + sr) * K + kn + sh * 16);
        rb0 = gb[0]; rb1 = gb[1];
      } else {
        const uint4* gb = (const uint4*)(Bb + (long long)(kn + bk) * N + bn0 + bn);
        rb0 = gb[0]; rb1 = gb[1];
      }
    }

    // fragments from LDS: 2x ds_read_b128 per fragment.
    // 16-bit A layout (ISA 7.12.2): lane-half h needs k in [8h,8h+7] then
    // [16+8h,16+8h+7] -> two contiguous 8-f16 (16 B) runs.
    v16h afr[2], bfr[4];
#pragma unroll
    for (int i = 0; i < 2; ++i) {
      const _Float16* p = &lA[cur][(wm * 32 + i * 16 + l16) * LDST + half * 8];
      ((v8h*)&afr[i])[0] = *(const v8h*)p;
      ((v8h*)&afr[i])[1] = *(const v8h*)(p + 16);
    }
#pragma unroll
    for (int j = 0; j < 4; ++j) {
      const _Float16* p = &lB[cur][(wn * 64 + j * 16 + l16) * LDST + half * 8];
      ((v8h*)&bfr[j])[0] = *(const v8h*)p;
      ((v8h*)&bfr[j])[1] = *(const v8h*)(p + 16);
    }
#pragma unroll
    for (int i = 0; i < 2; ++i)
#pragma unroll
      for (int j = 0; j < 4; ++j)
        acc[i][j] = __builtin_amdgcn_wmma_f32_16x16x32_f16(
            false, afr[i], false, bfr[j], (short)0, acc[i][j], false, false);

    if (has) {
      const int nxt = cur ^ 1;
      uint4* da = (uint4*)&lA[nxt][sr * LDST + sh * 16];
      da[0] = ra0; da[1] = ra1;
      if (TRANSB) {
        uint4* db = (uint4*)&lB[nxt][sr * LDST + sh * 16];
        db[0] = rb0; db[1] = rb1;
      } else {
        _Float16 t[16];
        *(uint4*)&t[0] = rb0;
        *(uint4*)&t[8] = rb1;
#pragma unroll
        for (int j = 0; j < 16; ++j) lB[nxt][(bn + j) * LDST + bk] = t[j];
      }
    }
    __syncthreads();
  }

  // ---- epilogue: C layout VGPR r -> M = r + 8*half, N = lane&15.
  // N is compile-time -> all store deltas are immediate offsets.
  const int row0 = bm0 + wm * 32 + half * 8;
  const int col0 = bn0 + wn * 64 + l16;
  float bv[4];
#pragma unroll
  for (int j = 0; j < 4; ++j) bv[j] = BIAS ? bias[col0 + j * 16] : 0.0f;

  const long long cbase = (long long)blockIdx.z * sC + (long long)row0 * N + col0;
  if (OUT32) {
    float* base = Cf + cbase;
#pragma unroll
    for (int i = 0; i < 2; ++i)
#pragma unroll
      for (int j = 0; j < 4; ++j)
#pragma unroll
        for (int r = 0; r < 8; ++r) {
          float vo = acc[i][j][r] * scale + bv[j];
          if (RELU) vo = fmaxf(vo, 0.0f);
          base[(i * 16 + r) * N + j * 16] = vo;
        }
  } else {
    _Float16* base = Ch + cbase;
#pragma unroll
    for (int i = 0; i < 2; ++i)
#pragma unroll
      for (int j = 0; j < 4; ++j)
#pragma unroll
        for (int r = 0; r < 8; ++r) {
          float vo = acc[i][j][r] * scale + bv[j];
          if (RELU) vo = fmaxf(vo, 0.0f);
          base[(i * 16 + r) * N + j * 16] = (_Float16)vo;
        }
  }
}

// y(f16) = x(f32) + sinusoidal positional encoding (computed on the fly).
__global__ __launch_bounds__(256) void add_pe_h(
    const float* __restrict__ x, _Float16* __restrict__ y,
    int S, int D, long long total) {
  long long i = (long long)blockIdx.x * blockDim.x + threadIdx.x;
  if (i >= total) return;
  const int d = (int)(i % D);
  const int s = (int)((i / D) % S);
  const float freq = __expf((float)(d & ~1) * (-__logf(10000.0f) / (float)D));
  const float ang  = (float)s * freq;
  const float pe   = (d & 1) ? __cosf(ang) : __sinf(ang);
  y[i] = (_Float16)(x[i] + pe);
}

// f32 -> f16 weight conversion (once per layer; 0.5 MB per matrix).
__global__ __launch_bounds__(256) void w_to_h(
    const float* __restrict__ w, _Float16* __restrict__ o, int n) {
  int i = blockIdx.x * blockDim.x + threadIdx.x;
  if (i < n) o[i] = (_Float16)w[i];
}

// Row softmax: read f32 scores, write f16 probabilities. 8-wave block per row.
__global__ __launch_bounds__(256) void softmax_rows(
    const float* __restrict__ S, _Float16* __restrict__ P, int ncols) {
  const float* r = S + (long long)blockIdx.x * ncols;
  _Float16*    o = P + (long long)blockIdx.x * ncols;
  const int tid = threadIdx.x, lane = tid & 31, wid = tid >> 5;
  __shared__ float sm[8];

  float m = -3.402823e38f;
  for (int c = tid; c < ncols; c += blockDim.x) m = fmaxf(m, r[c]);
  for (int off = 16; off; off >>= 1) m = fmaxf(m, __shfl_xor(m, off, 32));
  if (!lane) sm[wid] = m;
  __syncthreads();
  if (tid == 0) {
    float v = sm[0];
    for (int i = 1; i < 8; ++i) v = fmaxf(v, sm[i]);
    sm[0] = v;
  }
  __syncthreads();
  m = sm[0];
  __syncthreads();

  float s = 0.0f;
  for (int c = tid; c < ncols; c += blockDim.x) {
    const float e = __expf(r[c] - m);
    o[c] = (_Float16)e;          // e <= 1, safe in f16
    s += e;
  }
  for (int off = 16; off; off >>= 1) s += __shfl_xor(s, off, 32);
  if (!lane) sm[wid] = s;
  __syncthreads();
  if (tid == 0) {
    float v = 0.0f;
    for (int i = 0; i < 8; ++i) v += sm[i];
    sm[0] = v;
  }
  __syncthreads();
  const float inv = 1.0f / sm[0];
  for (int c = tid; c < ncols; c += blockDim.x)
    o[c] = (_Float16)((float)o[c] * inv);
}

// Final layer: out[b] = mean_s relu(X[b,s,:].wo + bo). X is f16, wo/bo f32.
__global__ __launch_bounds__(256) void head_mean(
    const _Float16* __restrict__ X, const float* __restrict__ wo,
    const float* __restrict__ bo, float* __restrict__ out, int S, int D) {
  const int b = blockIdx.x;
  const _Float16* Xb = X + (long long)b * S * D;
  const int tid = threadIdx.x, lane = tid & 31, wid = tid >> 5;
  __shared__ float sm[8];
  const float b0 = bo[0];

  float acc = 0.0f;
  for (int s = tid; s < S; s += blockDim.x) {
    const v8h* xr = (const v8h*)(Xb + (long long)s * D);
    float dot = b0;
    for (int d = 0; d < D / 8; ++d) {
      const v8h h = xr[d];
#pragma unroll
      for (int e = 0; e < 8; ++e) dot += (float)h[e] * wo[d * 8 + e];
    }
    acc += fmaxf(dot, 0.0f);
  }
  for (int off = 16; off; off >>= 1) acc += __shfl_xor(acc, off, 32);
  if (!lane) sm[wid] = acc;
  __syncthreads();
  if (tid == 0) {
    float v = 0.0f;
    for (int i = 0; i < 8; ++i) v += sm[i];
    out[b] = v / (float)S;
  }
}

extern "C" void kernel_launch(void* const* d_in, const int* in_sizes, int n_in,
                              void* d_out, int out_size, void* d_ws, size_t ws_size,
                              hipStream_t stream) {
  (void)in_sizes; (void)n_in; (void)out_size; (void)ws_size;
  const float* x = (const float*)d_in[0];
  // params flattened per layer: wq,bq,wk,bk,wv,bv,wo,bo
  auto P = [&](int layer, int idx) { return (const float*)d_in[1 + layer * 8 + idx]; };

  // ---- workspace carving (f16 activations, f32 scores) ------------------
  char* wp = (char*)d_ws;
  auto carve = [&](size_t bytes) {
    void* r = (void*)wp;
    wp += (bytes + 255) & ~(size_t)255;
    return r;
  };
  const size_t actB = (size_t)MROWS * D_MODEL * sizeof(_Float16);   // 32 MB
  _Float16* Xh  = (_Float16*)carve(actB);                 // layer input
  _Float16* Qh  = (_Float16*)carve(actB);                 // Q, then attn-out
  _Float16* Kh  = (_Float16*)carve(actB);
  _Float16* Vh  = (_Float16*)carve(actB);
  float*    SC  = (float*)   carve((size_t)NBATCH * SEQ * SEQ * 4); // 128 MB
  _Float16* Ph  = (_Float16*)carve((size_t)NBATCH * SEQ * SEQ * 2); //  64 MB
  _Float16* Wqh = (_Float16*)carve((size_t)D_MODEL * D_MODEL * 2);
  _Float16* Wkh = (_Float16*)carve((size_t)D_MODEL * D_MODEL * 2);
  _Float16* Wvh = (_Float16*)carve((size_t)D_MODEL * D_MODEL * 2);
  _Float16* Woh = (_Float16*)carve((size_t)D_MODEL * D_MODEL * 2);

  const dim3 blk(256);
  const float inv_sqrt_d = 1.0f / sqrtf((float)D_MODEL);
  const int WN = D_MODEL * D_MODEL;

  // x + positional encoding -> Xh (f16)
  {
    const long long total = (long long)MROWS * D_MODEL;
    add_pe_h<<<(unsigned)((total + 255) / 256), blk, 0, stream>>>(x, Xh, SEQ, D_MODEL, total);
  }

  const dim3 gProj(MROWS / BM, D_MODEL / BN, 1);   // 256 x 4
  const dim3 gScore(SEQ / BM, SEQ / BN, NBATCH);   // 8 x 8 x 32
  const dim3 gAV(SEQ / BM, D_MODEL / BN, NBATCH);  // 8 x 4 x 32
  const long long sQK = (long long)SEQ * D_MODEL;
  const long long sSC = (long long)SEQ * SEQ;

  for (int l = 0; l < 3; ++l) {
    const float *wq = P(l, 0), *bq = P(l, 1), *wk = P(l, 2), *bk = P(l, 3);
    const float *wv = P(l, 4), *bv = P(l, 5), *wo = P(l, 6), *bo = P(l, 7);

    // convert this layer's square weights to f16
    w_to_h<<<(WN + 255) / 256, blk, 0, stream>>>(wq, Wqh, WN);
    w_to_h<<<(WN + 255) / 256, blk, 0, stream>>>(wk, Wkh, WN);
    w_to_h<<<(WN + 255) / 256, blk, 0, stream>>>(wv, Wvh, WN);
    if (l < 2) w_to_h<<<(WN + 255) / 256, blk, 0, stream>>>(wo, Woh, WN);

    // Q/K/V projections over all (b, s) rows at once  (N=512, K=512)
    gemm_wmma<D_MODEL, D_MODEL, false, true, false, false><<<gProj, blk, 0, stream>>>(
        Xh, Wqh, bq, Qh, nullptr, 1.0f, 0, 0, 0);
    gemm_wmma<D_MODEL, D_MODEL, false, true, false, false><<<gProj, blk, 0, stream>>>(
        Xh, Wkh, bk, Kh, nullptr, 1.0f, 0, 0, 0);
    gemm_wmma<D_MODEL, D_MODEL, false, true, false, false><<<gProj, blk, 0, stream>>>(
        Xh, Wvh, bv, Vh, nullptr, 1.0f, 0, 0, 0);

    // scores (f32) = Q . K^T / sqrt(d)   (N=1024, K=512, B as N x K)
    gemm_wmma<SEQ, D_MODEL, true, false, false, true><<<gScore, blk, 0, stream>>>(
        Qh, Kh, nullptr, nullptr, SC, inv_sqrt_d, sQK, sQK, sSC);

    // softmax over keys -> f16 probabilities
    softmax_rows<<<NBATCH * SEQ, blk, 0, stream>>>(SC, Ph, SEQ);

    // attn_out (f16) = probs . V  (N=512, K=1024) -> reuse Qh (dead now)
    gemm_wmma<D_MODEL, SEQ, false, false, false, false><<<gAV, blk, 0, stream>>>(
        Ph, Vh, nullptr, Qh, nullptr, 1.0f, sSC, sQK, sQK);

    if (l < 2) {
      // x_next = relu(attn_out . wo + bo) -> Xh (not read by this GEMM)
      gemm_wmma<D_MODEL, D_MODEL, false, true, true, false><<<gProj, blk, 0, stream>>>(
          Qh, Woh, bo, Xh, nullptr, 1.0f, 0, 0, 0);
    } else {
      // final 512->1 projection + relu + mean over seq
      head_mean<<<NBATCH, blk, 0, stream>>>(Qh, wo, bo, (float*)d_out, SEQ, D_MODEL);
    }
  }
}
